// FEATURE_rec_16930761081280
// MI455X (gfx1250) — compile-verified, matrix operand order checked by reading, not run on hardware
//
#include <hip/hip_runtime.h>
#include <hip/hip_bf16.h>

// ---------------------------------------------------------------------------
// CDNA5 (gfx1250) wave32 WMMA implementation of a 2-layer masked graph
// attention network.  Matrix math: v_wmma_f32_16x16x32_f16.
// Flash-attention K/V chunks are staged to LDS with CDNA5 async loads
// (global_load_async_to_lds_b128, fenced by s_wait_asynccnt + barrier),
// double-buffered so the async engine overlaps the WMMA stream.
//
// WMMA 16x16x32 f16 register layouts (ISA 7.12.2):
//   A (16x32, MxK): lane l holds row M = l&15; base = 8*(l>>4):
//     halves a[0..7]  = K (base .. base+7), a[8..15] = K (16+base .. 16+base+7)
//   B (32x16, KxN): lane l holds row (l&15) of B^T, same K packing as A.
//   C/D (16x16 f32): lane l holds column N = l&15; acc[i] = row M = i+8*(l>>4).
//
// Score tile computed as S^T = K x Q^T: its D layout (lane = q column,
// VGPR i = key row) is exactly the A layout of the P x V WMMA, so softmaxed
// probabilities feed the PV matmul with no transpose.
// ---------------------------------------------------------------------------

typedef __attribute__((ext_vector_type(16))) _Float16     v16h;
typedef __attribute__((ext_vector_type(8)))  float        v8f;
typedef __attribute__((ext_vector_type(4)))  unsigned int v4u;
typedef __attribute__((ext_vector_type(4)))  float        v4f;

union AB { v16h v; v4u u[2]; };

#define BATCH 8
#define NSEQ  2048
#define DH    128
#define BN    (BATCH * NSEQ)
#define CHUNK 32
#define BIGC  9.0e15f

// ---------------------------------------------------------------------------
__global__ void cvt_f32_f16(const float* __restrict__ s, _Float16* __restrict__ d, int n) {
  int i = blockIdx.x * blockDim.x + threadIdx.x;
  if (i < n) d[i] = (_Float16)s[i];
}

// vae2_features -> columns [128,256) of the concat buffer [BN,256]
__global__ void cvt_vae_concat(const float* __restrict__ s, _Float16* __restrict__ d, int n) {
  int i = blockIdx.x * blockDim.x + threadIdx.x;
  if (i >= n) return;
  int r = i >> 7, c = i & 127;
  d[(size_t)r * 256 + 128 + c] = (_Float16)s[i];
}

// ---------------------------------------------------------------------------
// Generic WMMA GEMM: out[M,128] = act(X[M,K] @ W[128,K]^T + bias)
// Block = 128 threads (4 waves); wave -> one 16x16 output tile.
// ---------------------------------------------------------------------------
__global__ void __launch_bounds__(128)
gemm_wmma(const _Float16* __restrict__ X, int ldx, int K,
          const _Float16* __restrict__ W,
          const float* __restrict__ bias,
          _Float16* __restrict__ out16, float* __restrict__ out32,
          int ldo, int col_off, int relu, int trans_v)
{
  const int lane = threadIdx.x & 31;
  const int wv   = threadIdx.x >> 5;
  const int half = lane >> 4, col = lane & 15, base = half * 8;
  const int m0 = blockIdx.x * 16;
  const int n0 = (blockIdx.y * 4 + wv) * 16;

  const _Float16* xr = X + (size_t)(m0 + col) * ldx;
  const _Float16* wr = W + (size_t)(n0 + col) * K;

  v8f acc = {0.f, 0.f, 0.f, 0.f, 0.f, 0.f, 0.f, 0.f};
#pragma unroll
  for (int ks = 0; ks < K; ks += 32) {
    AB a, b;
    a.u[0] = *(const v4u*)(xr + ks + base);
    a.u[1] = *(const v4u*)(xr + ks + 16 + base);
    b.u[0] = *(const v4u*)(wr + ks + base);
    b.u[1] = *(const v4u*)(wr + ks + 16 + base);
    acc = __builtin_amdgcn_wmma_f32_16x16x32_f16(false, a.v, false, b.v,
                                                 (short)0, acc, false, false);
  }
  const float bb = bias[n0 + col];
#pragma unroll
  for (int i = 0; i < 8; ++i) {
    float r = acc[i] + bb;
    if (relu) r = fmaxf(r, 0.f);
    acc[i] = r;
  }
  if (out32) {
#pragma unroll
    for (int i = 0; i < 8; ++i)
      out32[(size_t)(m0 + i + 8 * half) * ldo + col_off + n0 + col] = acc[i];
  } else if (trans_v) {
#pragma unroll
    for (int i = 0; i < 8; ++i) {
      int r  = m0 + i + 8 * half;
      int b_ = r >> 11;            // row / NSEQ
      int n_ = r & (NSEQ - 1);
      out16[((size_t)b_ * DH + n0 + col) * NSEQ + n_] = (_Float16)acc[i];
    }
  } else {
#pragma unroll
    for (int i = 0; i < 8; ++i)
      out16[(size_t)(m0 + i + 8 * half) * ldo + col_off + n0 + col] = (_Float16)acc[i];
  }
}

// ---------------------------------------------------------------------------
// Async staging of one 32-key chunk of K (32x128 f16) and V^T (128x32 f16)
// into LDS, partitioned across all 128 threads of the block.
// Each thread issues 8x global_load_async_to_lds_b128 (ASYNCcnt-tracked).
// ---------------------------------------------------------------------------
__device__ __forceinline__ void async_stage_chunk(
    const _Float16* __restrict__ kg,          // k rows of this chunk (row-major [32][128])
    const _Float16* __restrict__ vg,          // vT base of this chunk  ([128][NSEQ], +c applied)
    _Float16 (*kb_)[DH], _Float16 (*vb_)[CHUNK], int tid)
{
#pragma unroll
  for (int j = 0; j < 4; ++j) {               // K: 512 16B segments / 128 threads
    int idx = tid + 128 * j;
    int row = idx >> 4, seg = idx & 15;
    unsigned long long g = (unsigned long long)(uintptr_t)(kg + row * DH + seg * 8);
    unsigned lds = (unsigned)(uintptr_t)&kb_[row][seg * 8];
    asm volatile("global_load_async_to_lds_b128 %0, %1, off"
                 :: "v"(lds), "v"(g) : "memory");
  }
#pragma unroll
  for (int j = 0; j < 4; ++j) {               // V^T: 512 16B segments / 128 threads
    int idx = tid + 128 * j;
    int f = idx >> 2, seg = idx & 3;
    unsigned long long g = (unsigned long long)(uintptr_t)(vg + (size_t)f * NSEQ + seg * 8);
    unsigned lds = (unsigned)(uintptr_t)&vb_[f][seg * 8];
    asm volatile("global_load_async_to_lds_b128 %0, %1, off"
                 :: "v"(lds), "v"(g) : "memory");
  }
}

// ---------------------------------------------------------------------------
// Fused masked attention with online softmax (flash style).
// One wave per (batch, 16-query) tile; 4 waves per block share the SAME
// batch, so K/V chunks are staged once per block into double-buffered LDS.
// adj is read exactly once; [N,N] scores never touch memory.
// ---------------------------------------------------------------------------
__global__ void __launch_bounds__(128)
flash_attn_wmma(const _Float16* __restrict__ q16, const _Float16* __restrict__ k16,
                const _Float16* __restrict__ vT16, const float* __restrict__ adj,
                _Float16* __restrict__ o16)
{
  __shared__ _Float16 kbuf[2][CHUNK][DH];     // 2 x 8KB
  __shared__ _Float16 vbuf[2][DH][CHUNK];     // 2 x 8KB

  const int tid  = threadIdx.x;
  const int lane = tid & 31;
  const int wv   = tid >> 5;
  const int tile = blockIdx.x * 4 + wv;
  const int b    = tile >> 7;                 // 128 q-tiles per batch; 4|128 ->
  const int qb   = (tile & 127) << 4;         // all 4 waves share batch b
  const int half = lane >> 4, col = lane & 15, base = half * 8;

  // Q tile (B operand of S^T): lane holds q row (l&15), persists in regs.
  const _Float16* qr = q16 + (size_t)(b * NSEQ + qb + col) * DH;
  AB qv[4];
#pragma unroll
  for (int s = 0; s < 4; ++s) {
    qv[s].u[0] = *(const v4u*)(qr + s * 32 + base);
    qv[s].u[1] = *(const v4u*)(qr + s * 32 + 16 + base);
  }
  const _Float16* kbg = k16  + (size_t)b * NSEQ * DH;
  const _Float16* vbg = vT16 + (size_t)b * DH * NSEQ;
  const float*    ar  = adj  + (size_t)(b * NSEQ + qb + col) * NSEQ;

  float m_run = -INFINITY, l_run = 0.f;
  const v8f VZ = {0.f, 0.f, 0.f, 0.f, 0.f, 0.f, 0.f, 0.f};
  v8f oacc[8];
#pragma unroll
  for (int t = 0; t < 8; ++t) oacc[t] = VZ;

  // prologue: stage chunk 0
  async_stage_chunk(kbg, vbg, kbuf[0], vbuf[0], tid);
  asm volatile("s_wait_asynccnt 0" ::: "memory");
  __syncthreads();
  int stage = 0;

#pragma unroll 1
  for (int c = 0; c < NSEQ; c += CHUNK) {
    // kick off async copies for the NEXT chunk into the other buffer
    if (c + CHUNK < NSEQ)
      async_stage_chunk(kbg + (size_t)(c + CHUNK) * DH, vbg + (c + CHUNK),
                        kbuf[stage ^ 1], vbuf[stage ^ 1], tid);

    _Float16 (*kc)[DH]    = kbuf[stage];
    _Float16 (*vc)[CHUNK] = vbuf[stage];

    // ---- S^T = K x Q^T, two 16-key subtiles, operands from LDS (ds_load_b128)
    v8f sc0 = VZ, sc1 = VZ;
#pragma unroll
    for (int s = 0; s < 4; ++s) {
      AB a;
      a.u[0] = *(const v4u*)&kc[col][s * 32 + base];
      a.u[1] = *(const v4u*)&kc[col][s * 32 + 16 + base];
      sc0 = __builtin_amdgcn_wmma_f32_16x16x32_f16(false, a.v, false, qv[s].v,
                                                   (short)0, sc0, false, false);
      AB a2;
      a2.u[0] = *(const v4u*)&kc[16 + col][s * 32 + base];
      a2.u[1] = *(const v4u*)&kc[16 + col][s * 32 + 16 + base];
      sc1 = __builtin_amdgcn_wmma_f32_16x16x32_f16(false, a2.v, false, qv[s].v,
                                                   (short)0, sc1, false, false);
    }

    // ---- mask: masked = s*adj - BIG*(1-adj)  (adj lanes match sc layout)
    float p[16];
    v4f a0 = *(const v4f*)(ar + c + base);
    v4f a1 = *(const v4f*)(ar + c + base + 4);
    v4f a2 = *(const v4f*)(ar + c + 16 + base);
    v4f a3 = *(const v4f*)(ar + c + 16 + base + 4);
#pragma unroll
    for (int i = 0; i < 4; ++i) {
      p[i]      = sc0[i]     * a0[i] - BIGC * (1.f - a0[i]);
      p[4 + i]  = sc0[4 + i] * a1[i] - BIGC * (1.f - a1[i]);
      p[8 + i]  = sc1[i]     * a2[i] - BIGC * (1.f - a2[i]);
      p[12 + i] = sc1[4 + i] * a3[i] - BIGC * (1.f - a3[i]);
    }

    // ---- online softmax (per q = lane&15; half-waves hold complementary keys)
    float cm = p[0];
#pragma unroll
    for (int e = 1; e < 16; ++e) cm = fmaxf(cm, p[e]);
    cm = fmaxf(cm, __shfl_xor(cm, 16, 32));
    const float m_new = fmaxf(m_run, cm);
    const float scale = __expf(m_run - m_new);

    AB pa;
    float lsum = 0.f;
#pragma unroll
    for (int e = 0; e < 16; ++e) {
      float pe = __expf(p[e] - m_new);
      lsum += pe;
      pa.v[e] = (_Float16)pe;                 // == A operand layout of P x V
    }
    l_run = l_run * scale + lsum;

    float srow[8];
#pragma unroll
    for (int i = 0; i < 8; ++i) srow[i] = __shfl(scale, i + 8 * half, 32);
#pragma unroll
    for (int t = 0; t < 8; ++t)
#pragma unroll
      for (int i = 0; i < 8; ++i) oacc[t][i] *= srow[i];

    // ---- O += P x V  (8 feature tiles; V^T rows contiguous in LDS)
#pragma unroll
    for (int t = 0; t < 8; ++t) {
      AB bv;
      bv.u[0] = *(const v4u*)&vc[t * 16 + col][base];
      bv.u[1] = *(const v4u*)&vc[t * 16 + col][16 + base];
      oacc[t] = __builtin_amdgcn_wmma_f32_16x16x32_f16(false, pa.v, false, bv.v,
                                                       (short)0, oacc[t], false, false);
    }
    m_run = m_new;

    if (c + CHUNK < NSEQ)
      __builtin_prefetch(ar + c + CHUNK, 0, 1);   // global_prefetch_b8 on adj

    // fence: next-chunk async copies complete, all waves done with this buffer
    asm volatile("s_wait_asynccnt 0" ::: "memory");
    __syncthreads();
    stage ^= 1;
  }

  // ---- normalize and store (f16, row-major for the following GEMM)
  float lt  = l_run + __shfl_xor(l_run, 16, 32);
  float inv = 1.0f / lt;
  float rinv[8];
#pragma unroll
  for (int i = 0; i < 8; ++i) rinv[i] = __shfl(inv, i + 8 * half, 32);
#pragma unroll
  for (int t = 0; t < 8; ++t)
#pragma unroll
    for (int i = 0; i < 8; ++i)
      o16[(size_t)(b * NSEQ + qb + i + 8 * half) * DH + t * 16 + col] =
          (_Float16)(oacc[t][i] * rinv[i]);
}

// ---------------------------------------------------------------------------
extern "C" void kernel_launch(void* const* d_in, const int* in_sizes, int n_in,
                              void* d_out, int out_size, void* d_ws, size_t ws_size,
                              hipStream_t stream) {
  (void)in_sizes; (void)n_in; (void)out_size; (void)ws_size;

  const float* h   = (const float*)d_in[0];
  const float* adj = (const float*)d_in[1];
  const float* vae = (const float*)d_in[2];
  const float* bv1 = (const float*)d_in[4];
  const float* bk1 = (const float*)d_in[6];
  const float* bq1 = (const float*)d_in[8];
  const float* bo1 = (const float*)d_in[10];
  const float* bv2 = (const float*)d_in[12];
  const float* bk2 = (const float*)d_in[14];
  const float* bq2 = (const float*)d_in[16];
  const float* bo2 = (const float*)d_in[18];
  const float* bf  = (const float*)d_in[20];

  char* ws = (char*)d_ws;
  size_t off = 0;
  auto alloc = [&](size_t bytes) -> void* {
    void* p = ws + off;
    off += (bytes + 255) & ~(size_t)255;
    return p;
  };

  _Float16* h16   = (_Float16*)alloc((size_t)BN * DH * 2);
  _Float16* q16   = (_Float16*)alloc((size_t)BN * DH * 2);
  _Float16* k16   = (_Float16*)alloc((size_t)BN * DH * 2);
  _Float16* vT16  = (_Float16*)alloc((size_t)BN * DH * 2);
  _Float16* o16   = (_Float16*)alloc((size_t)BN * DH * 2);
  _Float16* f116  = (_Float16*)alloc((size_t)BN * DH * 2);
  _Float16* cat16 = (_Float16*)alloc((size_t)BN * 256 * 2);

  // weight conversion: order = Wq1,Wk1,Wv1,Wo1,Wq2,Wk2,Wv2,Wo2,Wf
  const float* wsrc[9] = {
    (const float*)d_in[7],  (const float*)d_in[5],  (const float*)d_in[3],
    (const float*)d_in[9],  (const float*)d_in[15], (const float*)d_in[13],
    (const float*)d_in[11], (const float*)d_in[17], (const float*)d_in[19] };
  const int wn[9] = {16384,16384,16384,16384,16384,16384,16384,16384,32768};
  _Float16* w16[9];
  for (int i = 0; i < 9; ++i) {
    w16[i] = (_Float16*)alloc((size_t)wn[i] * 2);
    cvt_f32_f16<<<(wn[i] + 255) / 256, 256, 0, stream>>>(wsrc[i], w16[i], wn[i]);
  }

  const int nEl = BN * DH;
  cvt_f32_f16<<<nEl / 256, 256, 0, stream>>>(h, h16, nEl);
  cvt_vae_concat<<<nEl / 256, 256, 0, stream>>>(vae, cat16, nEl);

  dim3 gG(BN / 16, 2), bG(128);
  // ---- layer 1
  gemm_wmma<<<gG, bG, 0, stream>>>(h16, DH, DH, w16[0], bq1, q16,  nullptr, DH, 0, 1, 0);
  gemm_wmma<<<gG, bG, 0, stream>>>(h16, DH, DH, w16[1], bk1, k16,  nullptr, DH, 0, 1, 0);
  gemm_wmma<<<gG, bG, 0, stream>>>(h16, DH, DH, w16[2], bv1, vT16, nullptr, DH, 0, 1, 1);
  flash_attn_wmma<<<BN / 16 / 4, 128, 0, stream>>>(q16, k16, vT16, adj, o16);
  gemm_wmma<<<gG, bG, 0, stream>>>(o16, DH, DH, w16[3], bo1, f116, nullptr, DH, 0, 0, 0);
  // ---- layer 2
  gemm_wmma<<<gG, bG, 0, stream>>>(f116, DH, DH, w16[4], bq2, q16,  nullptr, DH, 0, 1, 0);
  gemm_wmma<<<gG, bG, 0, stream>>>(f116, DH, DH, w16[5], bk2, k16,  nullptr, DH, 0, 1, 0);
  gemm_wmma<<<gG, bG, 0, stream>>>(f116, DH, DH, w16[6], bv2, vT16, nullptr, DH, 0, 1, 1);
  flash_attn_wmma<<<BN / 16 / 4, 128, 0, stream>>>(q16, k16, vT16, adj, o16);
  // f2 -> columns [0,128) of concat buffer
  gemm_wmma<<<gG, bG, 0, stream>>>(o16, DH, DH, w16[7], bo2, cat16, nullptr, 256, 0, 0, 0);
  // ---- final: [BN,256] @ Wf^T -> d_out (f32)
  gemm_wmma<<<gG, bG, 0, stream>>>(cat16, 256, 256, w16[8], bf, nullptr,
                                   (float*)d_out, DH, 0, 0, 0);
}
